// SSM_spa_6528350290350
// MI455X (gfx1250) — matrix operand order repaired and checked by reading, not compiled, not run on hardware
//
#include <hip/hip_runtime.h>
#include <hip/hip_bf16.h>

// ---------------------------------------------------------------------------
// Mamba SSM block forward for MI455X (gfx1250, wave32, WMMA + TDM).
// GEMMs: operands pre-converted to bf16, V_WMMA_F32_16X16X32_BF16 compute,
// fp32 accumulate. Tiles DMA'd into LDS via TENSOR_LOAD_TO_LDS (double
// buffered, TENSORcnt-synchronized) when the builtin is available.
// ---------------------------------------------------------------------------

typedef __bf16        bf16x16 __attribute__((ext_vector_type(16)));
typedef float         f32x8   __attribute__((ext_vector_type(8)));
typedef float         f32x4   __attribute__((ext_vector_type(4)));
typedef unsigned int  u32x4   __attribute__((ext_vector_type(4)));
typedef unsigned int  u32x2   __attribute__((ext_vector_type(2)));
typedef int           i32x4   __attribute__((ext_vector_type(4)));
typedef int           i32x8   __attribute__((ext_vector_type(8)));

union Frag { u32x4 q[2]; bf16x16 v; };

__device__ __forceinline__ unsigned short f2bf(float f) {
    unsigned int u = __float_as_uint(f);
    unsigned int r = u + 0x7fffu + ((u >> 16) & 1u);   // RNE to bf16
    return (unsigned short)(r >> 16);
}
__device__ __forceinline__ unsigned int pk2(float lo, float hi) {
    return (unsigned int)f2bf(lo) | ((unsigned int)f2bf(hi) << 16);
}
__device__ __forceinline__ float softplusf(float x) {
    return (x > 20.f) ? x : log1pf(__expf(x));
}
__device__ __forceinline__ float sigmoidf_(float x) {
    return 1.f / (1.f + __expf(-x));
}

#if __has_builtin(__builtin_amdgcn_tensor_load_to_lds) && \
    __has_builtin(__builtin_amdgcn_s_wait_tensorcnt)
#define HAVE_TDM 1
#else
#define HAVE_TDM 0
#endif

#if HAVE_TDM
// Issue one TDM load of a 128-row x 32-col bf16 tile from a row-major
// [rows x ld] bf16 tensor into LDS. LDS rows are padded 32 data shorts +
// 16 pad shorts (pad_interval = 16 DWORDs, pad_amount = 8 DWORDs) to match
// the 48-short LDS row stride used by the WMMA fragment readers.
// One instruction = one DMA (EXEC ignored); call from a single wave.
__device__ __forceinline__ void tdm_load_tile(const unsigned short* base,
                                              int ld, int row0, int k0,
                                              unsigned int lds_addr,
                                              int nrows) {
    unsigned long long ga =
        (unsigned long long)(base + (size_t)row0 * ld + k0);
    u32x4 g0;
    g0.x = 1u;                                        // count=1, user mode
    g0.y = lds_addr;                                  // lds_addr[31:0]
    g0.z = (unsigned int)ga;                          // global_addr[31:0]
    g0.w = (unsigned int)((ga >> 32) & 0x1FFFFFFull)  // global_addr[56:32]
           | (2u << 30);                              // type = 2 (image)
    i32x8 g1;
    // data_size=1 (2B) | pad_enable | pad_interval=3 (16 DW) | pad_amount=7 (8 DW)
    g1[0] = (int)((1u << 16) | (1u << 20) | (3u << 22) | (7u << 25));
    unsigned int td0 = (unsigned int)ld;              // tensor_dim0 (elems)
    unsigned int td1 = (unsigned int)nrows;           // tensor_dim1 (rows)
    g1[1] = (int)((td0 & 0xFFFFu) << 16);             // tensor_dim0[15:0]
    g1[2] = (int)((td0 >> 16) | ((td1 & 0xFFFFu) << 16));
    g1[3] = (int)((td1 >> 16) | (32u << 16));         // tile_dim0 = 32
    g1[4] = (int)128u;                                // tile_dim1=128, tile_dim2=0
    unsigned long long s0 = (unsigned long long)ld;   // tensor_dim0_stride
    g1[5] = (int)(s0 & 0xFFFFFFFFull);
    g1[6] = (int)(s0 >> 32);                          // dim1_stride = 0
    g1[7] = 0;
    i32x4 z4 = {0, 0, 0, 0};
#if defined(__clang_major__) && (__clang_major__ >= 23)
    i32x8 z8 = {0, 0, 0, 0, 0, 0, 0, 0};
    __builtin_amdgcn_tensor_load_to_lds(g0, g1, z4, z4, z8, 0);
#else
    __builtin_amdgcn_tensor_load_to_lds(g0, g1, z4, z4, 0);
#endif
}
#endif  // HAVE_TDM

// C[m,n] = sum_k A[m*lda+k] * Bw[n*ldb+k]  (bf16 in, fp32 out, opt. softplus)
// Block tile 128x128x32, double-buffered LDS; 8 waves in 4(M) x 2(N);
// each wave: 32x64 region = 2x4 WMMA tiles (64 accum VGPRs/lane).
__global__ __launch_bounds__(256) void gemm_bf16_wmma(
    const unsigned short* __restrict__ A, const unsigned short* __restrict__ Bw,
    float* __restrict__ C, int M, int Npad, int K,
    int lda, int ldb, int ldc, const float* __restrict__ bias, int mode) {
    __shared__ unsigned short As[2][128][48];
    __shared__ unsigned short Bs[2][128][48];

    const int tid  = threadIdx.x;
    const int lane = tid & 31;
    const int wave = tid >> 5;
    const int wm   = wave >> 1;          // 0..3 : 32-row stripe
    const int wn   = wave & 1;           // 0..1 : 64-col stripe
    const int hv   = lane >> 4;
    const int lid  = lane & 15;

    const int nTiles = Npad >> 7;
    const int m0 = (blockIdx.x / nTiles) << 7;
    const int n0 = (blockIdx.x % nTiles) << 7;

    f32x8 acc[2][4];
#pragma unroll
    for (int i = 0; i < 2; ++i)
#pragma unroll
        for (int j = 0; j < 4; ++j)
#pragma unroll
            for (int r = 0; r < 8; ++r) acc[i][j][r] = 0.f;

    auto stage = [&](int bufi, int k0_) {
#if HAVE_TDM
        if (wave == 0)
            tdm_load_tile(A, lda, m0, k0_,
                          (unsigned int)(unsigned long long)&As[bufi][0][0], M);
        else if (wave == 1)
            tdm_load_tile(Bw, ldb, n0, k0_,
                          (unsigned int)(unsigned long long)&Bs[bufi][0][0],
                          Npad);
#else
        const int row = tid >> 1;
        const int kh  = (tid & 1) << 4;
        const u32x4* pa = reinterpret_cast<const u32x4*>(
            A + (size_t)(m0 + row) * lda + k0_ + kh);
        u32x4 a0 = pa[0], a1 = pa[1];
        u32x4* da = reinterpret_cast<u32x4*>(&As[bufi][row][kh]);
        da[0] = a0; da[1] = a1;
        const u32x4* pb = reinterpret_cast<const u32x4*>(
            Bw + (size_t)(n0 + row) * ldb + k0_ + kh);
        u32x4 b0 = pb[0], b1 = pb[1];
        u32x4* db = reinterpret_cast<u32x4*>(&Bs[bufi][row][kh]);
        db[0] = b0; db[1] = b1;
        if (k0_ + 32 < K) {  // global_prefetch_b8 toward next K-tile
            __builtin_prefetch(A + (size_t)(m0 + row) * lda + k0_ + 32 + kh, 0, 0);
            __builtin_prefetch(Bw + (size_t)(n0 + row) * ldb + k0_ + 32 + kh, 0, 0);
        }
#endif
    };
    auto stage_sync = [&]() {
#if HAVE_TDM
        if (wave <= 1) __builtin_amdgcn_s_wait_tensorcnt(0);
#endif
        __syncthreads();
    };

    stage(0, 0);
    stage_sync();
    int buf = 0;
    for (int k0 = 0; k0 < K; k0 += 32) {
        const int nxt = k0 + 32;
        if (nxt < K) stage(buf ^ 1, nxt);   // DMA next tile while computing

        Frag fa[2], fb[4];
        // A fragment (16x32 bf16): e0-7 -> K=hv*8+0..7, e8-15 -> K=16+hv*8+0..7
#pragma unroll
        for (int i = 0; i < 2; ++i) {
            const unsigned short* r = &As[buf][(wm << 5) + (i << 4) + lid][0];
            fa[i].q[0] = *reinterpret_cast<const u32x4*>(r + hv * 8);
            fa[i].q[1] = *reinterpret_cast<const u32x4*>(r + 16 + hv * 8);
        }
        // B fragment (32x16 bf16): n = lid, e0-7 -> K=hv*8+0..7, e8-15 -> K=16+hv*8+0..7
#pragma unroll
        for (int j = 0; j < 4; ++j) {
            const unsigned short* r = &Bs[buf][(wn << 6) + (j << 4) + lid][0];
            fb[j].q[0] = *reinterpret_cast<const u32x4*>(r + hv * 8);
            fb[j].q[1] = *reinterpret_cast<const u32x4*>(r + 16 + hv * 8);
        }
#pragma unroll
        for (int i = 0; i < 2; ++i)
#pragma unroll
            for (int j = 0; j < 4; ++j)
                acc[i][j] = __builtin_amdgcn_wmma_f32_16x16x32_bf16(
                    false, fa[i].v, false, fb[j].v, (short)0, acc[i][j],
                    false, false);
        if (nxt < K) stage_sync();
        buf ^= 1;
    }

    // C/D layout: lane l, reg r -> m = r + (l>>4)*8, n = l&15
#pragma unroll
    for (int i = 0; i < 2; ++i)
#pragma unroll
        for (int j = 0; j < 4; ++j) {
            int gn = n0 + (wn << 6) + (j << 4) + lid;
#pragma unroll
            for (int r = 0; r < 8; ++r) {
                int gm = m0 + (wm << 5) + (i << 4) + hv * 8 + r;
                float v = acc[i][j][r];
                if (mode == 1) v = softplusf(v + bias[gn]);
                C[(size_t)gm * ldc + gn] = v;
            }
        }
}

// fp32 -> bf16 (RNE), 4 elements/thread, with optional row padding:
// rows >= src_rows produce zeros (used to pad x_proj_w 96 -> 128 rows).
__global__ __launch_bounds__(256) void cvt_bf16_pad(
    const float* __restrict__ src, unsigned short* __restrict__ dst,
    int rows, int cols, int src_rows, int src_ld) {
    int i4 = (blockIdx.x * 256 + threadIdx.x) * 4;
    if (i4 >= rows * cols) return;
    int r = i4 / cols;
    int c = i4 - r * cols;
    u32x2 o;
    if (r < src_rows) {
        f32x4 f = *reinterpret_cast<const f32x4*>(src + (size_t)r * src_ld + c);
        o.x = pk2(f.x, f.y);
        o.y = pk2(f.z, f.w);
    } else {
        o.x = 0u; o.y = 0u;
    }
    *reinterpret_cast<u32x2*>(dst + i4) = o;
}

// Depthwise causal conv1d (k=4, left pad 3) + bias + SiLU.
// Input xs_pre = XZ[:, 0:2048] (row stride 4096). Writes fp32 (for scan)
// and bf16 (x_proj GEMM operand).
__global__ void conv_silu_kernel(const float* __restrict__ XZ,
                                 const float* __restrict__ cw,
                                 const float* __restrict__ cb,
                                 float* __restrict__ xs,
                                 unsigned short* __restrict__ xsbf) {
    int idx = blockIdx.x * blockDim.x + threadIdx.x;   // B*L*2048
    int d = idx & 2047;
    int t = (idx >> 11) & 255;
    int b = idx >> 19;
    float acc = cb[d];
#pragma unroll
    for (int j = 0; j < 4; ++j) {
        int t2 = t - 3 + j;
        if (t2 >= 0)
            acc += cw[d * 4 + j] * XZ[((size_t)(b * 256 + t2)) * 4096 + d];
    }
    float v = acc * sigmoidf_(acc);
    xs[idx] = v;
    xsbf[idx] = f2bf(v);
}

// Selective scan: one thread per (b, d); N=16 states in registers.
// Per-batch B/C matrices (256x16 each) staged in 32 KB LDS.
// Fuses y = (scan + D*xs) * silu(res); writes bf16 (out_proj GEMM operand).
__global__ __launch_bounds__(256) void scan_kernel(
    const float* __restrict__ xdbl,   // (B*L,128): [0:64)=dt [64:80)=B [80:96)=C
    const float* __restrict__ delta,  // (B*L,2048) post-softplus
    const float* __restrict__ xs,     // (B*L,2048)
    const float* __restrict__ A_log,  // (2048,16)
    const float* __restrict__ Dp,     // (2048)
    const float* __restrict__ XZ,     // (B*L,4096), res = cols [2048:4096)
    unsigned short* __restrict__ yb) {  // (B*L,2048) bf16
    __shared__ float BmS[256 * 16];
    __shared__ float CmS[256 * 16];
    int b = blockIdx.x >> 3;
    int d = ((blockIdx.x & 7) << 8) + threadIdx.x;

    for (int i = threadIdx.x; i < 4096; i += 256) {
        int t = i >> 4, n = i & 15;
        const float* row = xdbl + (size_t)(b * 256 + t) * 128;
        BmS[i] = row[64 + n];
        CmS[i] = row[80 + n];
    }
    __syncthreads();

    float Areg[16], h[16];
#pragma unroll
    for (int n = 0; n < 16; ++n) {
        Areg[n] = -__expf(A_log[d * 16 + n]);
        h[n] = 0.f;
    }
    float Dd = Dp[d];

    for (int t = 0; t < 256; ++t) {
        size_t rb = (size_t)(b * 256 + t);
        float dlt = delta[rb * 2048 + d];
        float xsv = xs[rb * 2048 + d];
        float dbu = dlt * xsv;
        float accv = 0.f;
#pragma unroll
        for (int n = 0; n < 16; ++n) {
            float dA = __expf(dlt * Areg[n]);
            h[n] = dA * h[n] + dbu * BmS[t * 16 + n];
            accv += h[n] * CmS[t * 16 + n];
        }
        float yvv  = accv + Dd * xsv;
        float resv = XZ[rb * 4096 + 2048 + d];
        yb[rb * 2048 + d] = f2bf(yvv * (resv * sigmoidf_(resv)));
    }
}

// LayerNorm over last dim (1024); one block per row.
__global__ __launch_bounds__(256) void ln_kernel(const float* __restrict__ yo,
                                                 const float* __restrict__ w,
                                                 const float* __restrict__ bia,
                                                 float* __restrict__ out) {
    __shared__ float rs[256], rq[256];
    int row = blockIdx.x;
    int tid = threadIdx.x;
    const float* p = yo + (size_t)row * 1024;
    float s = 0.f, sq = 0.f;
    for (int i = tid; i < 1024; i += 256) {
        float v = p[i];
        s += v; sq += v * v;
    }
    rs[tid] = s; rq[tid] = sq;
    __syncthreads();
    for (int off = 128; off > 0; off >>= 1) {
        if (tid < off) { rs[tid] += rs[tid + off]; rq[tid] += rq[tid + off]; }
        __syncthreads();
    }
    float mu  = rs[0] * (1.f / 1024.f);
    float var = rq[0] * (1.f / 1024.f) - mu * mu;
    float inv = rsqrtf(var + 1e-5f);
    for (int i = tid; i < 1024; i += 256)
        out[(size_t)row * 1024 + i] = (p[i] - mu) * inv * w[i] + bia[i];
}

extern "C" void kernel_launch(void* const* d_in, const int* in_sizes, int n_in,
                              void* d_out, int out_size, void* d_ws,
                              size_t ws_size, hipStream_t stream) {
    const float* x1   = (const float*)d_in[0];   // (4,256,32,32) == (1024,1024)
    const float* Wi   = (const float*)d_in[1];   // (4096,1024)
    const float* cw   = (const float*)d_in[2];   // (2048,1,4)
    const float* cb   = (const float*)d_in[3];   // (2048)
    const float* Wx   = (const float*)d_in[4];   // (96,2048)
    const float* Wdt  = (const float*)d_in[5];   // (2048,64)
    const float* bdt  = (const float*)d_in[6];   // (2048)
    const float* Alog = (const float*)d_in[7];   // (2048,16)
    const float* Dp   = (const float*)d_in[8];   // (2048)
    const float* Wo   = (const float*)d_in[9];   // (1024,2048)
    const float* lnw  = (const float*)d_in[10];  // (1024)
    const float* lnb  = (const float*)d_in[11];  // (1024)
    float* out = (float*)d_out;

    char* w = (char*)d_ws;
    float*          XZ    = (float*)w;          w += (size_t)1024 * 4096 * 4;
    float*          xs    = (float*)w;          w += (size_t)1024 * 2048 * 4;
    float*          xdbl  = (float*)w;          w += (size_t)1024 * 128  * 4;
    float*          delta = (float*)w;          w += (size_t)1024 * 2048 * 4;
    float*          yo    = (float*)w;          w += (size_t)1024 * 1024 * 4;
    unsigned short* Xbf   = (unsigned short*)w; w += (size_t)1024 * 1024 * 2;
    unsigned short* Wibf  = (unsigned short*)w; w += (size_t)4096 * 1024 * 2;
    unsigned short* Wxbf  = (unsigned short*)w; w += (size_t)128  * 2048 * 2;
    unsigned short* Wdtbf = (unsigned short*)w; w += (size_t)2048 * 64   * 2;
    unsigned short* Wobf  = (unsigned short*)w; w += (size_t)1024 * 2048 * 2;
    unsigned short* xsbf  = (unsigned short*)w; w += (size_t)1024 * 2048 * 2;
    unsigned short* dtbf  = (unsigned short*)w; w += (size_t)1024 * 64   * 2;
    unsigned short* ybf   = (unsigned short*)w; w += (size_t)1024 * 2048 * 2;

    // 0) one-time bf16 conversions (weights + input)
    cvt_bf16_pad<<<dim3(1024), 256, 0, stream>>>(x1, Xbf, 1024, 1024, 1024, 1024);
    cvt_bf16_pad<<<dim3(4096), 256, 0, stream>>>(Wi, Wibf, 4096, 1024, 4096, 1024);
    cvt_bf16_pad<<<dim3(256),  256, 0, stream>>>(Wx, Wxbf, 128, 2048, 96, 2048);
    cvt_bf16_pad<<<dim3(128),  256, 0, stream>>>(Wdt, Wdtbf, 2048, 64, 2048, 64);
    cvt_bf16_pad<<<dim3(2048), 256, 0, stream>>>(Wo, Wobf, 1024, 2048, 1024, 2048);

    // 1) in_proj: (1024,1024) x (4096,1024)^T -> XZ (1024,4096)
    gemm_bf16_wmma<<<dim3(8 * 32), 256, 0, stream>>>(
        Xbf, Wibf, XZ, 1024, 4096, 1024, 1024, 1024, 4096, nullptr, 0);
    // 2) depthwise causal conv + SiLU -> xs (f32 + bf16)
    conv_silu_kernel<<<dim3((4 * 256 * 2048) / 256), 256, 0, stream>>>(
        XZ, cw, cb, xs, xsbf);
    // 3) x_proj: (1024,2048) x (96,2048)^T -> xdbl (1024,128 padded)
    gemm_bf16_wmma<<<dim3(8 * 1), 256, 0, stream>>>(
        xsbf, Wxbf, xdbl, 1024, 128, 2048, 2048, 2048, 128, nullptr, 0);
    // 3b) dt columns -> bf16 (1024,64)
    cvt_bf16_pad<<<dim3(64), 256, 0, stream>>>(xdbl, dtbf, 1024, 64, 1024, 128);
    // 4) dt_proj + bias + softplus: (1024,64) x (2048,64)^T -> delta
    gemm_bf16_wmma<<<dim3(8 * 16), 256, 0, stream>>>(
        dtbf, Wdtbf, delta, 1024, 2048, 64, 64, 64, 2048, bdt, 1);
    // 5) selective scan + D*xs + silu(res) gate -> ybf (bf16)
    scan_kernel<<<dim3(32), 256, 0, stream>>>(xdbl, delta, xs, Alog, Dp, XZ, ybf);
    // 6) out_proj: (1024,2048) x (1024,2048)^T -> yo
    gemm_bf16_wmma<<<dim3(8 * 8), 256, 0, stream>>>(
        ybf, Wobf, yo, 1024, 1024, 2048, 2048, 2048, 1024, nullptr, 0);
    // 7) LayerNorm -> out
    ln_kernel<<<dim3(1024), 256, 0, stream>>>(yo, lnw, lnb, out);
}
